// MyTransformer_47665547051847
// MI455X (gfx1250) — matrix operand order repaired and checked by reading, not compiled, not run on hardware
//
#include <hip/hip_runtime.h>
#include <hip/hip_bf16.h>
#include <math.h>

// ---------------------------------------------------------------------------
// Transformer encoder-decoder for MI455X (gfx1250).
// All matmuls on the WMMA matrix pipe: v_wmma_f32_16x16x32_bf16
// (fp32 -> bf16 conversion fused into LDS staging, f32 accumulation).
// LDS tiles are double-buffered and stored pre-swizzled into WMMA fragment
// order so each lane loads its fragment with contiguous ds_load_b128s.
// Global staging uses float4 (global_load_b128) wherever alignment allows.
// ---------------------------------------------------------------------------

#define D_MODEL 512
#define N_HEADS 8
#define D_HEAD  64
#define D_FF    2048
#define ENC_LEN 2048
#define DEC_LEN 1024
#define VOCAB   50257
#define N_LAYERS 6

typedef __attribute__((ext_vector_type(16))) __bf16 v16bf;
typedef __attribute__((ext_vector_type(8)))  float  v8f;

static __device__ __forceinline__ unsigned short f2bf_raw(float f) {
  unsigned u = __builtin_bit_cast(unsigned, f);
  unsigned r = u + 0x7FFFu + ((u >> 16) & 1u);   // round-to-nearest-even
  return (unsigned short)(r >> 16);
}
static __device__ __forceinline__ __bf16 f2bf(float f) {
  unsigned short h = f2bf_raw(f);
  return __builtin_bit_cast(__bf16, h);
}
// Pack two floats into two bf16 (low = x, high = y). Uses the HW packed
// converter when the toolchain exposes it, else manual RNE.
static __device__ __forceinline__ unsigned pack2_bf16(float x, float y) {
#if defined(__has_builtin)
#if __has_builtin(__builtin_amdgcn_cvt_pk_bf16_f32)
  auto pk = __builtin_amdgcn_cvt_pk_bf16_f32(x, y);
  unsigned u;
  __builtin_memcpy(&u, &pk, 4);
  return u;
#else
  return (unsigned)f2bf_raw(x) | ((unsigned)f2bf_raw(y) << 16);
#endif
#else
  return (unsigned)f2bf_raw(x) | ((unsigned)f2bf_raw(y) << 16);
#endif
}

// ---------------------------------------------------------------------------
// GEMM: C[M,N] = A[M,K] * B[K,N]   (row-major fp32 in/out, bf16 WMMA inside)
// Requires M % 128 == 0, K % 32 == 0 (true for all calls). N may be ragged.
//
// Block: 256 threads = 8 waves; block tile 128x128; BK = 32 (= WMMA K).
// Waves 4(M) x 2(N); each wave owns 32x64: 2 A-frags x 4 B-frags -> 8 WMMAs.
//
// Fragment-ordered LDS [buf][tile][lane][elem] (bf16):
//  A map (16x32): lane = (r&15)|16*bit4(k), elem = k&15
//  B map (32x16): lane = (n&15)|16*bit4(k), elem = k&15
// Double-buffered: one barrier per K-step; next tile's global loads overlap
// the current tile's WMMAs.
// ---------------------------------------------------------------------------
#define BM 128
#define BN 128
#define BK 32

__global__ __launch_bounds__(256)
void gemm_bf16_wmma(const float* __restrict__ A, const float* __restrict__ B,
                    float* __restrict__ C, int M, int N, int K, int relu) {
  __shared__ __align__(32) __bf16 Afrag[2][8][32][16];   // 16 KB
  __shared__ __align__(32) __bf16 Bfrag[2][8][32][16];   // 16 KB

  const int tid  = threadIdx.x;
  const int lane = tid & 31;
  const int wave = tid >> 5;           // 0..7
  const int wm   = wave >> 1;          // 0..3 -> rows wm*32 .. wm*32+31
  const int wn   = wave & 1;           // 0..1 -> cols wn*64 .. wn*64+63
  const int row0 = blockIdx.y * BM;
  const int col0 = blockIdx.x * BN;
  const bool nvec = (N & 3) == 0;      // float4 path legal for B

  v8f acc[2][4];
#pragma unroll
  for (int mi = 0; mi < 2; ++mi)
#pragma unroll
    for (int ni = 0; ni < 4; ++ni)
#pragma unroll
      for (int r = 0; r < 8; ++r) acc[mi][ni][r] = 0.0f;

  float4 aReg[4];        // A: 128x32 tile  = 1024 float4, 4 per thread
  float4 bRegV[4];       // B: 32x128 tile  = 1024 float4, 4 per thread
  float  bRegS[16];      // B scalar fallback (ragged N)

  auto loadTiles = [&](int kt) {
    const float* Ag = A + (size_t)row0 * K + (size_t)kt * BK;
#pragma unroll
    for (int j = 0; j < 4; ++j) {
      int f = j * 256 + tid;
      int r = f >> 3, c4 = f & 7;                       // row 0..127, float4-col 0..7
      aReg[j] = *reinterpret_cast<const float4*>(Ag + (size_t)r * K + c4 * 4);
    }
    const float* Bg = B + (size_t)(kt * BK) * N + col0;
    if (nvec) {
#pragma unroll
      for (int j = 0; j < 4; ++j) {
        int f = j * 256 + tid;
        int r = f >> 5, c = (f & 31) * 4;               // k 0..31, n 0..127 (x4)
        if (col0 + c < N) {
          bRegV[j] = *reinterpret_cast<const float4*>(Bg + (size_t)r * N + c);
        } else {
          bRegV[j] = make_float4(0.f, 0.f, 0.f, 0.f);
        }
      }
    } else {
#pragma unroll
      for (int j = 0; j < 16; ++j) {
        int li = j * 256 + tid;
        int r = li >> 7, c = li & 127;                  // k 0..31, n 0..127
        bRegS[j] = (col0 + c < N) ? Bg[(size_t)r * N + c] : 0.0f;
      }
    }
  };

  auto storeTiles = [&](int buf) {
#pragma unroll
    for (int j = 0; j < 4; ++j) {                       // A: packed ds_store_b64
      int f = j * 256 + tid;
      int r = f >> 3, c = (f & 7) * 4;
      int mt = r >> 4;
      int fl = (r & 15) | (c & 16);
      int fe = c & 15;                                  // in {0,4,8,12}: 8B aligned
      uint2 pk;
      pk.x = pack2_bf16(aReg[j].x, aReg[j].y);
      pk.y = pack2_bf16(aReg[j].z, aReg[j].w);
      *reinterpret_cast<uint2*>(&Afrag[buf][mt][fl][fe]) = pk;
    }
    if (nvec) {
#pragma unroll
      for (int j = 0; j < 4; ++j) {                     // B: 4 lanes, same elem
        int f = j * 256 + tid;
        int r = f >> 5, c = (f & 31) * 4;
        int nt = c >> 4;
        int fl = (c & 15) | (r & 16);
        int fe = r & 15;
        Bfrag[buf][nt][fl + 0][fe] = f2bf(bRegV[j].x);
        Bfrag[buf][nt][fl + 1][fe] = f2bf(bRegV[j].y);
        Bfrag[buf][nt][fl + 2][fe] = f2bf(bRegV[j].z);
        Bfrag[buf][nt][fl + 3][fe] = f2bf(bRegV[j].w);
      }
    } else {
#pragma unroll
      for (int j = 0; j < 16; ++j) {
        int li = j * 256 + tid;
        int r = li >> 7, c = li & 127;
        int nt = c >> 4;
        int fl = (c & 15) | (r & 16);
        int fe = r & 15;
        Bfrag[buf][nt][fl][fe] = f2bf(bRegS[j]);
      }
    }
  };

  const int steps = K / BK;
  loadTiles(0);
  storeTiles(0);

  for (int kt = 0; kt < steps; ++kt) {
    const int cur = kt & 1;
    __syncthreads();                       // buf[cur] ready; prior reads of buf[cur^1] done
    const bool more = (kt + 1) < steps;
    if (more) loadTiles(kt + 1);           // global loads overlap WMMAs below

    v16bf a[2], b[4];
#pragma unroll
    for (int mi = 0; mi < 2; ++mi)
      a[mi] = *reinterpret_cast<const v16bf*>(&Afrag[cur][wm * 2 + mi][lane][0]);
#pragma unroll
    for (int ni = 0; ni < 4; ++ni)
      b[ni] = *reinterpret_cast<const v16bf*>(&Bfrag[cur][wn * 4 + ni][lane][0]);

#pragma unroll
    for (int mi = 0; mi < 2; ++mi)
#pragma unroll
      for (int ni = 0; ni < 4; ++ni)
        acc[mi][ni] = __builtin_amdgcn_wmma_f32_16x16x32_bf16(
            false, a[mi], false, b[ni], (short)0, acc[mi][ni], false, false);

    if (more) storeTiles(cur ^ 1);
  }

  // Epilogue. C/D layout: elem r -> row = r + 8*(lane>=16), col = lane&15
  const int rsub = ((lane & 16) ? 8 : 0);
  const int csub = lane & 15;
#pragma unroll
  for (int mi = 0; mi < 2; ++mi) {
#pragma unroll
    for (int ni = 0; ni < 4; ++ni) {
      int rbase = row0 + (wm * 2 + mi) * 16 + rsub;
      int col   = col0 + (wn * 4 + ni) * 16 + csub;
      if (col < N) {
#pragma unroll
        for (int r = 0; r < 8; ++r) {
          float v = acc[mi][ni][r];
          if (relu) v = fmaxf(v, 0.0f);
          C[(size_t)(rbase + r) * N + col] = v;
        }
      }
    }
  }
}

// ---------------------------------------------------------------------------
// Flash-style attention: one wave per (query, head).
// QKV buffers are [S, 1536]: cols [0,512)=Q, [512,1024)=K, [1024,1536)=V.
// Causal mask matches the reference: score += -1e9 for k > q.
// ---------------------------------------------------------------------------
__global__ __launch_bounds__(32)
void attention_kernel(const float* __restrict__ Qb, const float* __restrict__ KVb,
                      float* __restrict__ Out, int Sq, int Sk, int causal) {
  __shared__ float qs[D_HEAD];
  const int q = blockIdx.x;
  const int h = blockIdx.y;
  const int lane = threadIdx.x;  // 0..31

  const float* qrow = Qb + (size_t)q * (3 * D_MODEL) + h * D_HEAD;
  qs[lane]      = qrow[lane];
  qs[lane + 32] = qrow[lane + 32];
  __syncthreads();

  float m = -3.0e38f, lsum = 0.0f, acc0 = 0.0f, acc1 = 0.0f;

  for (int kb = 0; kb < Sk; kb += 32) {
    const int kidx = kb + lane;
    const float* krow = KVb + (size_t)kidx * (3 * D_MODEL) + D_MODEL + h * D_HEAD;
    float s = 0.0f;
#pragma unroll
    for (int d = 0; d < D_HEAD; ++d) s += qs[d] * krow[d];
    s *= 0.125f;                       // 1/sqrt(64)
    if (causal && kidx > q) s -= 1.0e9f;

    float cmax = s;
    for (int off = 16; off > 0; off >>= 1)
      cmax = fmaxf(cmax, __shfl_xor(cmax, off, 32));
    const float mnew = fmaxf(m, cmax);
    const float scale = __expf(m - mnew);
    acc0 *= scale; acc1 *= scale; lsum *= scale;

    const float p = __expf(s - mnew);
    float psum = p;
    for (int off = 16; off > 0; off >>= 1) psum += __shfl_xor(psum, off, 32);
    lsum += psum;
    m = mnew;

    const float* vbase = KVb + (size_t)kb * (3 * D_MODEL) + 2 * D_MODEL + h * D_HEAD;
#pragma unroll 4
    for (int j = 0; j < 32; ++j) {
      const float pj = __shfl(p, j, 32);
      const float* vrow = vbase + (size_t)j * (3 * D_MODEL);
      acc0 += pj * vrow[lane];
      acc1 += pj * vrow[lane + 32];
    }
  }

  const float inv = 1.0f / lsum;
  float* orow = Out + (size_t)q * D_MODEL + h * D_HEAD;
  orow[lane]      = acc0 * inv;
  orow[lane + 32] = acc1 * inv;
}

// ---------------------------------------------------------------------------
// add & layernorm (no affine), one 128-thread block per row of 512.
// In-place safe: each thread only rewrites its own elements.
// ---------------------------------------------------------------------------
__global__ __launch_bounds__(128)
void add_norm_kernel(const float* __restrict__ X, const float* __restrict__ Y,
                     float* __restrict__ O) {
  __shared__ float s1[128], s2[128];
  const int row = blockIdx.x;
  const int t = threadIdx.x;
  float v[4], sum = 0.0f, sq = 0.0f;
#pragma unroll
  for (int j = 0; j < 4; ++j) {
    int idx = t + j * 128;
    float a = X[(size_t)row * D_MODEL + idx] + Y[(size_t)row * D_MODEL + idx];
    v[j] = a; sum += a; sq += a * a;
  }
  s1[t] = sum; s2[t] = sq;
  __syncthreads();
  for (int off = 64; off > 0; off >>= 1) {
    if (t < off) { s1[t] += s1[t + off]; s2[t] += s2[t + off]; }
    __syncthreads();
  }
  const float mean = s1[0] * (1.0f / D_MODEL);
  const float var  = s2[0] * (1.0f / D_MODEL) - mean * mean;
  const float rstd = rsqrtf(var + 1e-6f);
#pragma unroll
  for (int j = 0; j < 4; ++j) {
    int idx = t + j * 128;
    O[(size_t)row * D_MODEL + idx] = (v[j] - mean) * rstd;
  }
}

// ---------------------------------------------------------------------------
// Embedding gather + sinusoidal positional encoding (start_pos = 0).
// ---------------------------------------------------------------------------
__global__ __launch_bounds__(256)
void embed_pe_kernel(const float* __restrict__ dict, const int* __restrict__ ids,
                     float* __restrict__ O) {
  const int row = blockIdx.x;
  const int t = threadIdx.x;
  const int id = ids[row];
  const float pos = (float)row;
#pragma unroll
  for (int j = 0; j < 2; ++j) {
    int i = t + j * 256;
    int base = i & ~1;
    float e = (float)base * (1.0f / D_MODEL);
    float ang = pos * expf(-9.210340371976184f * e);   // pos / 10000^e
    float pe = (i & 1) ? cosf(ang) : sinf(ang);
    O[(size_t)row * D_MODEL + i] = dict[(size_t)id * D_MODEL + i] + pe;
  }
}

// ---------------------------------------------------------------------------
// Host orchestration
// ---------------------------------------------------------------------------
static inline void launch_gemm(const float* A, const float* B, float* C,
                               int M, int N, int K, int relu, hipStream_t s) {
  dim3 grid((N + BN - 1) / BN, M / BM);
  gemm_bf16_wmma<<<grid, dim3(256), 0, s>>>(A, B, C, M, N, K, relu);
}

extern "C" void kernel_launch(void* const* d_in, const int* in_sizes, int n_in,
                              void* d_out, int out_size, void* d_ws, size_t ws_size,
                              hipStream_t stream) {
  (void)in_sizes; (void)n_in; (void)out_size; (void)ws_size;

  const int*   input_ids   = (const int*)  d_in[0];
  const int*   dec_ids     = (const int*)  d_in[1];
  const float* dict        = (const float*)d_in[2];
  const float* mh_input    = (const float*)d_in[3];
  const float* wo_input    = (const float*)d_in[4];
  const float* mh_masked   = (const float*)d_in[5];
  const float* wo_masked   = (const float*)d_in[6];
  const float* mh_output   = (const float*)d_in[7];
  const float* wo_output   = (const float*)d_in[8];
  const float* ff_in_w1    = (const float*)d_in[9];
  const float* ff_in_w2    = (const float*)d_in[10];
  const float* ff_out_w1   = (const float*)d_in[11];
  const float* ff_out_w2   = (const float*)d_in[12];
  const float* last_linear = (const float*)d_in[13];

  const size_t MHW = (size_t)D_MODEL * 3 * D_MODEL;   // 786432
  const size_t WOW = (size_t)D_MODEL * D_MODEL;       // 262144
  const size_t FFW = (size_t)D_MODEL * D_FF;          // 1048576

  // Workspace layout (fp32)
  float* ws    = (float*)d_ws;
  float* enc   = ws;                                   // 2048*512
  float* dec   = enc   + (size_t)ENC_LEN * D_MODEL;    // 1024*512
  float* qkvE  = dec   + (size_t)DEC_LEN * D_MODEL;    // 2048*1536
  float* qkvD  = qkvE  + (size_t)ENC_LEN * 3 * D_MODEL;// 1024*1536
  float* ctx   = qkvD  + (size_t)DEC_LEN * 3 * D_MODEL;// 2048*512
  float* proj  = ctx   + (size_t)ENC_LEN * D_MODEL;    // 2048*512
  float* ffh   = proj  + (size_t)ENC_LEN * D_MODEL;    // 2048*2048
  float* ffo   = ffh   + (size_t)ENC_LEN * D_FF;       // 2048*512

  // ----- Encoder -----
  embed_pe_kernel<<<ENC_LEN, 256, 0, stream>>>(dict, input_ids, enc);
  for (int i = 0; i < N_LAYERS; ++i) {
    launch_gemm(enc, mh_input + (size_t)i * MHW, qkvE, ENC_LEN, 3 * D_MODEL, D_MODEL, 0, stream);
    attention_kernel<<<dim3(ENC_LEN, N_HEADS), 32, 0, stream>>>(qkvE, qkvE, ctx, ENC_LEN, ENC_LEN, 0);
    launch_gemm(ctx, wo_input + (size_t)i * WOW, proj, ENC_LEN, D_MODEL, D_MODEL, 0, stream);
    add_norm_kernel<<<ENC_LEN, 128, 0, stream>>>(enc, proj, enc);
    launch_gemm(enc, ff_in_w1 + (size_t)i * FFW, ffh, ENC_LEN, D_FF, D_MODEL, 1, stream);
    launch_gemm(ffh, ff_in_w2 + (size_t)i * FFW, ffo, ENC_LEN, D_MODEL, D_FF, 0, stream);
    add_norm_kernel<<<ENC_LEN, 128, 0, stream>>>(enc, ffo, enc);
  }

  // ----- Decoder -----
  embed_pe_kernel<<<DEC_LEN, 256, 0, stream>>>(dict, dec_ids, dec);
  for (int i = 0; i < N_LAYERS; ++i) {
    // masked self-attention
    launch_gemm(dec, mh_masked + (size_t)i * MHW, qkvD, DEC_LEN, 3 * D_MODEL, D_MODEL, 0, stream);
    attention_kernel<<<dim3(DEC_LEN, N_HEADS), 32, 0, stream>>>(qkvD, qkvD, ctx, DEC_LEN, DEC_LEN, 1);
    launch_gemm(ctx, wo_masked + (size_t)i * WOW, proj, DEC_LEN, D_MODEL, D_MODEL, 0, stream);
    add_norm_kernel<<<DEC_LEN, 128, 0, stream>>>(dec, proj, dec);
    // cross-attention: Q from dec, K/V from enc (same weight block)
    launch_gemm(dec, mh_output + (size_t)i * MHW, qkvD, DEC_LEN, 3 * D_MODEL, D_MODEL, 0, stream);
    launch_gemm(enc, mh_output + (size_t)i * MHW, qkvE, ENC_LEN, 3 * D_MODEL, D_MODEL, 0, stream);
    attention_kernel<<<dim3(DEC_LEN, N_HEADS), 32, 0, stream>>>(qkvD, qkvE, ctx, DEC_LEN, ENC_LEN, 0);
    launch_gemm(ctx, wo_output + (size_t)i * WOW, proj, DEC_LEN, D_MODEL, D_MODEL, 0, stream);
    add_norm_kernel<<<DEC_LEN, 128, 0, stream>>>(dec, proj, dec);
    // FFN
    launch_gemm(dec, ff_out_w1 + (size_t)i * FFW, ffh, DEC_LEN, D_FF, D_MODEL, 1, stream);
    launch_gemm(ffh, ff_out_w2 + (size_t)i * FFW, ffo, DEC_LEN, D_MODEL, D_FF, 0, stream);
    add_norm_kernel<<<DEC_LEN, 128, 0, stream>>>(dec, ffo, dec);
  }

  // ----- Final projection to logits -----
  launch_gemm(dec, last_linear, (float*)d_out, DEC_LEN, VOCAB, D_MODEL, 0, stream);
}
